// Cross_54477365182468
// MI455X (gfx1250) — matrix operand order repaired and checked by reading, not compiled, not run on hardware
//
#include <hip/hip_runtime.h>
#include <hip/hip_bf16.h>

// ---------------------------------------------------------------------------
// Cross-attention (B=1, N=4096, C=256, H=8, d=32), bf16 WMMA + TDM on gfx1250.
// ---------------------------------------------------------------------------

typedef __attribute__((ext_vector_type(16))) __bf16 v16bf;
typedef __attribute__((ext_vector_type(8)))  float  v8f;
typedef __attribute__((ext_vector_type(4)))  unsigned int u32x4;
typedef __attribute__((ext_vector_type(8)))  int  i32x8;
typedef __attribute__((ext_vector_type(4)))  int  i32x4;

#define NTOK  4096
#define CDIM  256
#define HEADS 8
#define DHEAD 32
#define OC    (3 * CDIM)   // 768 qkv outputs per token

static_assert(sizeof(v16bf) == 32, "v16bf must be 32 bytes");

union BF16Frag { uint4 u4[2]; v16bf v; };

// Load a 16-element bf16 fragment from two 16-byte-aligned chunks (global or LDS).
__device__ __forceinline__ v16bf ld_frag(const __bf16* p0, const __bf16* p1) {
    BF16Frag f;
    f.u4[0] = *(const uint4*)p0;
    f.u4[1] = *(const uint4*)p1;
    return f.v;
}

// Convert 16 f32 (two 8-float runs) to a bf16 fragment.
__device__ __forceinline__ v16bf cvt_16f(const float* p0, const float* p1) {
    float4 a = *(const float4*)p0;
    float4 b = *(const float4*)(p0 + 4);
    float4 c = *(const float4*)p1;
    float4 d = *(const float4*)(p1 + 4);
    v16bf r;
    r[0] = (__bf16)a.x; r[1] = (__bf16)a.y; r[2] = (__bf16)a.z; r[3] = (__bf16)a.w;
    r[4] = (__bf16)b.x; r[5] = (__bf16)b.y; r[6] = (__bf16)b.z; r[7] = (__bf16)b.w;
    r[8] = (__bf16)c.x; r[9] = (__bf16)c.y; r[10] = (__bf16)c.z; r[11] = (__bf16)c.w;
    r[12] = (__bf16)d.x; r[13] = (__bf16)d.y; r[14] = (__bf16)d.z; r[15] = (__bf16)d.w;
    return r;
}

// Half-wave (16-lane) reductions: rows of a WMMA C/D tile live in half-waves.
__device__ __forceinline__ float half_max16(float v) {
    v = fmaxf(v, __shfl_xor(v, 1, 32));
    v = fmaxf(v, __shfl_xor(v, 2, 32));
    v = fmaxf(v, __shfl_xor(v, 4, 32));
    v = fmaxf(v, __shfl_xor(v, 8, 32));
    return v;
}
__device__ __forceinline__ float half_sum16(float v) {
    v += __shfl_xor(v, 1, 32);
    v += __shfl_xor(v, 2, 32);
    v += __shfl_xor(v, 4, 32);
    v += __shfl_xor(v, 8, 32);
    return v;
}
__device__ __forceinline__ float wave_sum32(float v) {
    v += __shfl_xor(v, 1, 32);
    v += __shfl_xor(v, 2, 32);
    v += __shfl_xor(v, 4, 32);
    v += __shfl_xor(v, 8, 32);
    v += __shfl_xor(v, 16, 32);
    return v;
}

// ---------------------------------------------------------------------------
// Tensor Data Mover: 2-D tile load (global -> LDS), D# per CDNA5 ISA ch.8.
// data_size_code: 0=1B 1=2B 2=4B. Dims/strides in data_size units.
// 6-arg builtin (clang-23 lane): (g0, g1, g2, g3, g4, cpol).
// Issue from ONE wave only; completion via s_wait_tensorcnt + barrier.
// ---------------------------------------------------------------------------
__device__ __forceinline__ void tdm_load_2d(unsigned lds_addr, const void* gptr,
                                            unsigned data_size_code,
                                            unsigned tile0, unsigned tile1,
                                            unsigned dim0, unsigned dim1,
                                            unsigned long long stride0) {
    unsigned long long ga = (unsigned long long)(uintptr_t)gptr;
    u32x4 g0;
    g0[0] = 1u;                                         // count=1, user mode
    g0[1] = lds_addr;                                   // LDS byte address
    g0[2] = (unsigned)(ga & 0xffffffffull);             // global_addr[31:0]
    g0[3] = (unsigned)((ga >> 32) & 0x01ffffffull)      // global_addr[56:32]
          | (2u << 30);                                 // type = 2 ("image")
    i32x8 g1;
    g1[0] = (int)(data_size_code << 16);                // wg_mask=0, data_size
    g1[1] = (int)((dim0 & 0xffffu) << 16);              // tensor_dim0[15:0]
    g1[2] = (int)(((dim0 >> 16) & 0xffffu) | ((dim1 & 0xffffu) << 16));
    g1[3] = (int)(((dim1 >> 16) & 0xffffu) | ((tile0 & 0xffffu) << 16));
    g1[4] = (int)(tile1 & 0xffffu);                     // tile_dim1; tile_dim2=0
    g1[5] = (int)(stride0 & 0xffffffffull);             // dim0_stride[31:0]
    g1[6] = (int)((stride0 >> 32) & 0xffffull);         // dim0_stride[47:32]
    g1[7] = 0;
    i32x4 z4 = {0, 0, 0, 0};                            // groups 2/3 unused (2-D)
    i32x8 z8 = {0, 0, 0, 0, 0, 0, 0, 0};                // group 4 unused
    __builtin_amdgcn_tensor_load_to_lds(g0, g1, z4, z4, z8, 0);
}

__device__ __forceinline__ unsigned lds_off(const void* p) {
    return (unsigned)(uintptr_t)p;   // low 32 bits of generic addr = LDS offset
}

// ---------------------------------------------------------------------------
// Kernel 1: QKV projection  t[src][n][o] = sum_c x[src][n][c] * W[o][c]
// X tile (16 rows x 256 = 16KB) staged by TDM; 8 waves x 6 o-tiles each.
// ---------------------------------------------------------------------------
__global__ void __launch_bounds__(256)
qkv_gemm_kernel(const float* __restrict__ before,
                const float* __restrict__ after,
                const float* __restrict__ W,
                float* __restrict__ tbuf) {
    __shared__ __align__(16) float xt[16][256];
    const int src   = blockIdx.y;
    const int nbase = blockIdx.x * 16;
    const float* X  = (src == 0) ? before : after;

    const int lane = threadIdx.x & 31;
    const int wid  = threadIdx.x >> 5;
    const int m    = lane & 15;
    const int hh   = lane >> 4;

    if (wid == 0) {   // 16 consecutive token rows: one contiguous 16KB 1-D tile
        tdm_load_2d(lds_off(&xt[0][0]), X + (size_t)nbase * CDIM,
                    /*4B*/2, /*tile*/16 * CDIM, 1, /*dims*/16 * CDIM, 1,
                    /*stride*/16 * CDIM);
        __builtin_amdgcn_s_wait_tensorcnt(0);
    }
    __syncthreads();

    for (int ot = 0; ot < 6; ++ot) {
        const int obase = (wid * 6 + ot) * 16;
        v8f acc = {};
#pragma unroll
        for (int kk = 0; kk < CDIM; kk += 32) {
            const float* xr = &xt[m][kk];
            v16bf a = cvt_16f(xr + 16 * hh, xr + 16 * hh + 8);
            const float* wp = W + (size_t)(obase + m) * CDIM + kk + 16 * hh;
            v16bf b = cvt_16f(wp, wp + 8);
            acc = __builtin_amdgcn_wmma_f32_16x16x32_bf16(
                false, a, false, b, (short)0, acc, false, false);
        }
        float* op = tbuf + ((size_t)src * NTOK + nbase) * OC + obase;
#pragma unroll
        for (int r = 0; r < 8; ++r)
            op[(size_t)(r + 8 * hh) * OC + m] = acc[r];
    }
}

// ---------------------------------------------------------------------------
// Kernel 2: per-head LayerNorm over d=32 (one wave per segment) + pack bf16.
// Q,K -> [src][h][n][d]; V -> transposed [src][h][d][n].
// ---------------------------------------------------------------------------
__global__ void __launch_bounds__(256)
ln_pack_kernel(const float* __restrict__ tbuf,
               const float* __restrict__ gamma,
               const float* __restrict__ beta,
               __bf16* __restrict__ qb,
               __bf16* __restrict__ kb,
               __bf16* __restrict__ vt) {
    const int lane = threadIdx.x & 31;
    const int wid  = threadIdx.x >> 5;
    const int seg  = blockIdx.x * 8 + wid;        // 2 * 4096 * 24 segments
    const int src  = seg / (NTOK * 24);
    const int rem  = seg - src * (NTOK * 24);
    const int n    = rem / 24;
    const int j    = rem - n * 24;                // j = qkv*8 + h
    const int qkv  = j >> 3;
    const int h    = j & 7;

    const float x  = tbuf[((size_t)src * NTOK + n) * OC + j * DHEAD + lane];
    const float mu = wave_sum32(x) * (1.0f / 32.0f);
    const float dx = x - mu;
    const float var = wave_sum32(dx * dx) * (1.0f / 32.0f);
    const float y  = dx * rsqrtf(var + 1e-5f) * gamma[lane] + beta[lane];
    const __bf16 yb = (__bf16)y;

    const size_t sh = (size_t)src * HEADS + h;
    if (qkv == 0)      qb[(sh * NTOK + n) * DHEAD + lane] = yb;
    else if (qkv == 1) kb[(sh * NTOK + n) * DHEAD + lane] = yb;
    else               vt[(sh * DHEAD + lane) * NTOK + n] = yb;   // transposed
}

// ---------------------------------------------------------------------------
// Kernel 3: flash attention with block-shared, TDM-staged, double-buffered K/V.
// Block = 8 waves, one (dir, head); wave owns 32 queries (two 16-row tiles).
// Per 32-key chunk: K tile [key][d] and V^T tile [d][key] land in LDS once,
// every wave runs 8 WMMAs against them while the next chunk DMAs in.
// ---------------------------------------------------------------------------
struct AttnSmem {
    __bf16 kt[2][32][32];   // [buf][key][d]   2 x 2KB
    __bf16 vtile[2][32][32];// [buf][d][key]   2 x 2KB
    __bf16 p[8][16][32];    // per-wave P      8KB
};

__global__ void __launch_bounds__(256)
attn_kernel(const __bf16* __restrict__ qb,
            const __bf16* __restrict__ kb,
            const __bf16* __restrict__ vt,
            float* __restrict__ out) {
    __shared__ __align__(16) AttnSmem sm;

    const int lane = threadIdx.x & 31;
    const int wid  = threadIdx.x >> 5;
    const int bx   = blockIdx.x;          // ((dir*8 + h)*16 + qc)
    const int qc   = bx & 15;
    const int dh   = bx >> 4;
    const int h    = dh & 7;
    const int dir  = dh >> 3;
    // dir 0: context_b = attn(q_after, k_before, v_before); dir 1: context_a.
    const int qsrc = (dir == 0) ? 1 : 0;
    const int ksrc = 1 - qsrc;
    const int m    = lane & 15;
    const int hh   = lane >> 4;
    const float scale = 0.17677669529663687f;  // 1/sqrt(32)

    const __bf16* qp = qb + (((size_t)qsrc * HEADS + h) * NTOK
                             + (size_t)qc * 256 + wid * 32) * DHEAD;
    const __bf16* kp = kb + (((size_t)ksrc * HEADS + h) * NTOK) * DHEAD;
    const __bf16* vp = vt + ((size_t)ksrc * HEADS + h) * (size_t)DHEAD * NTOK;

    // Q A-fragments for the wave's two 16-row tiles (held in regs all loop).
    v16bf aq[2];
#pragma unroll
    for (int t = 0; t < 2; ++t) {
        const __bf16* qrow = qp + (size_t)(t * 16 + m) * DHEAD;
        aq[t] = ld_frag(qrow + 16 * hh, qrow + 16 * hh + 8);
    }

    v8f oacc[2][2] = {};
    float mr[2][8], lr[2][8];
#pragma unroll
    for (int t = 0; t < 2; ++t)
#pragma unroll
        for (int r = 0; r < 8; ++r) { mr[t][r] = -3.0e38f; lr[t][r] = 0.0f; }

    // Prologue: DMA chunk 0 into buffer 0.
    if (wid == 0) {
        tdm_load_2d(lds_off(&sm.kt[0][0][0]), kp, 1, DHEAD, 32, DHEAD, NTOK, DHEAD);
        tdm_load_2d(lds_off(&sm.vtile[0][0][0]), vp, 1, 32, DHEAD, NTOK, DHEAD, NTOK);
        __builtin_amdgcn_s_wait_tensorcnt(0);
    }
    __syncthreads();

    for (int i = 0; i < NTOK / 32; ++i) {
        const int buf = i & 1;
        if (wid == 0 && i + 1 < NTOK / 32) {   // overlap next chunk's DMA
            const int kn = (i + 1) * 32;
            tdm_load_2d(lds_off(&sm.kt[buf ^ 1][0][0]), kp + (size_t)kn * DHEAD,
                        1, DHEAD, 32, DHEAD, NTOK, DHEAD);
            tdm_load_2d(lds_off(&sm.vtile[buf ^ 1][0][0]), vp + kn,
                        1, 32, DHEAD, NTOK, DHEAD, NTOK);
        }

        // Shared B-fragments for this chunk (K cols 0-15/16-31, V dd 0-15/16-31).
        const __bf16* k0 = &sm.kt[buf][m][16 * hh];
        const __bf16* k1 = &sm.kt[buf][16 + m][16 * hh];
        v16bf bk0 = ld_frag(k0, k0 + 8);
        v16bf bk1 = ld_frag(k1, k1 + 8);
        const __bf16* v0 = &sm.vtile[buf][m][16 * hh];
        const __bf16* v1 = &sm.vtile[buf][16 + m][16 * hh];
        v16bf bv0 = ld_frag(v0, v0 + 8);
        v16bf bv1 = ld_frag(v1, v1 + 8);

#pragma unroll
        for (int t = 0; t < 2; ++t) {
            v8f s0 = {}, s1 = {};
            s0 = __builtin_amdgcn_wmma_f32_16x16x32_bf16(false, aq[t], false, bk0,
                                                         (short)0, s0, false, false);
            s1 = __builtin_amdgcn_wmma_f32_16x16x32_bf16(false, aq[t], false, bk1,
                                                         (short)0, s1, false, false);
            // Online softmax per row (row r + 8*hh lives in one half-wave).
#pragma unroll
            for (int r = 0; r < 8; ++r) {
                const float a0 = s0[r] * scale;
                const float a1 = s1[r] * scale;
                const float mnew = fmaxf(mr[t][r], half_max16(fmaxf(a0, a1)));
                const float alpha = __expf(mr[t][r] - mnew);
                const float p0 = __expf(a0 - mnew);
                const float p1 = __expf(a1 - mnew);
                lr[t][r] = lr[t][r] * alpha + half_sum16(p0 + p1);
                mr[t][r] = mnew;
                oacc[t][0][r] *= alpha;
                oacc[t][1][r] *= alpha;
                const int row = r + 8 * hh;
                sm.p[wid][row][m]      = (__bf16)p0;
                sm.p[wid][row][16 + m] = (__bf16)p1;
            }
            // P: D-layout -> A-layout via the wave's private LDS slab.
            v16bf ap = ld_frag(&sm.p[wid][m][16 * hh], &sm.p[wid][m][16 * hh + 8]);
            oacc[t][0] = __builtin_amdgcn_wmma_f32_16x16x32_bf16(
                false, ap, false, bv0, (short)0, oacc[t][0], false, false);
            oacc[t][1] = __builtin_amdgcn_wmma_f32_16x16x32_bf16(
                false, ap, false, bv1, (short)0, oacc[t][1], false, false);
        }

        if (wid == 0) __builtin_amdgcn_s_wait_tensorcnt(0);
        __syncthreads();   // next chunk ready AND all waves done with `buf`
    }

    // Finalize: O / l, write [N, C] f32.
    float* ob = out + (size_t)dir * NTOK * CDIM
              + ((size_t)qc * 256 + wid * 32) * CDIM + h * DHEAD;
#pragma unroll
    for (int t = 0; t < 2; ++t)
#pragma unroll
        for (int r = 0; r < 8; ++r) {
            const float inv = 1.0f / lr[t][r];
            const int row = t * 16 + r + 8 * hh;
            ob[(size_t)row * CDIM + m]      = oacc[t][0][r] * inv;
            ob[(size_t)row * CDIM + 16 + m] = oacc[t][1][r] * inv;
        }
}

// ---------------------------------------------------------------------------
extern "C" void kernel_launch(void* const* d_in, const int* in_sizes, int n_in,
                              void* d_out, int out_size, void* d_ws, size_t ws_size,
                              hipStream_t stream) {
    const float* before = (const float*)d_in[0];
    const float* after  = (const float*)d_in[1];
    const float* W      = (const float*)d_in[2];
    const float* gamma  = (const float*)d_in[3];
    const float* beta   = (const float*)d_in[4];
    float* out = (float*)d_out;

    char* ws = (char*)d_ws;
    float*  tbuf = (float*)ws;                                        // 2*4096*768 f32
    __bf16* qbuf = (__bf16*)(ws + (size_t)2 * NTOK * OC * sizeof(float));
    __bf16* kbuf = qbuf + (size_t)2 * HEADS * NTOK * DHEAD;
    __bf16* vbuf = kbuf + (size_t)2 * HEADS * NTOK * DHEAD;

    qkv_gemm_kernel<<<dim3(NTOK / 16, 2), 256, 0, stream>>>(before, after, W, tbuf);
    ln_pack_kernel<<<(2 * NTOK * 24) / 8, 256, 0, stream>>>(tbuf, gamma, beta,
                                                            qbuf, kbuf, vbuf);
    attn_kernel<<<2 * HEADS * (NTOK / 256), 256, 0, stream>>>(qbuf, kbuf, vbuf, out);
}
